// Model_39676907885130
// MI455X (gfx1250) — matrix operand order repaired and checked by reading, not compiled, not run on hardware
//
#include <hip/hip_runtime.h>
#include <math.h>

// 512 (l,h) pairs, each: S = QK^T (512^3), column-softmax (axis=2), O = PV.
// Softmax folded into per-column constants c_k = m_k + log(Z_k), so
// O[i,j] = sum_k exp(scale*QK[i,k] - c_k) * V[k,j].
// Workspace: 512*512 floats (1 MB) for c.

#define SDIM   512
#define SS     (SDIM * SDIM)
#define SCALE  0.044194173824159216f   // 1/sqrt(512)
#define LDP    (SDIM + 8)              // padded bf16 LDS row stride
#define VTP    40                      // padded row stride for Vt chunk (32 + 8)

typedef __attribute__((ext_vector_type(16))) __bf16 bf16x16;
typedef __attribute__((ext_vector_type(8)))  float  f32x8;

__device__ __forceinline__ f32x8 zero8() {
  f32x8 z;
#pragma unroll
  for (int i = 0; i < 8; ++i) z[i] = 0.0f;
  return z;
}

// ---- A-operand fragment (16x32 bf16). Lane = row (mod 16); lanes 0-15 hold
// K 0..7 & 16..23, lanes 16-31 hold K 8..15 & 24..31. ----
__device__ __forceinline__ bf16x16 fragA_g32(const float* __restrict__ base,
                                             int ld, int lane) {
  const int row  = lane & 15;
  const int koff = (lane >> 4) << 3;          // 0 or 8
  const float* p = base + row * ld + koff;
  float4 c0 = *reinterpret_cast<const float4*>(p);
  float4 c1 = *reinterpret_cast<const float4*>(p + 4);
  float4 c2 = *reinterpret_cast<const float4*>(p + 16);
  float4 c3 = *reinterpret_cast<const float4*>(p + 20);
  bf16x16 a;
  a[0]  = (__bf16)c0.x; a[1]  = (__bf16)c0.y; a[2]  = (__bf16)c0.z; a[3]  = (__bf16)c0.w;
  a[4]  = (__bf16)c1.x; a[5]  = (__bf16)c1.y; a[6]  = (__bf16)c1.z; a[7]  = (__bf16)c1.w;
  a[8]  = (__bf16)c2.x; a[9]  = (__bf16)c2.y; a[10] = (__bf16)c2.z; a[11] = (__bf16)c2.w;
  a[12] = (__bf16)c3.x; a[13] = (__bf16)c3.y; a[14] = (__bf16)c3.z; a[15] = (__bf16)c3.w;
  return a;
}

__device__ __forceinline__ bf16x16 fragA_lds(const __bf16* base, int ld, int lane) {
  const int row  = lane & 15;
  const int koff = (lane >> 4) << 3;
  const __bf16* p = base + row * ld + koff;
  union { uint4 q[2]; bf16x16 v; } u;
  u.q[0] = *reinterpret_cast<const uint4*>(p);        // K koff..koff+7
  u.q[1] = *reinterpret_cast<const uint4*>(p + 16);   // K koff+16..koff+23
  return u.v;
}

// ---- B-operand fragment (32x16 bf16). Lane = output column (mod 16);
// lanes 0-15 hold K 0..7 & 16..23, lanes 16-31 hold K 8..15 & 24..31. ----
__device__ __forceinline__ bf16x16 fragB_g32(const float* __restrict__ base,
                                             int ld, int lane) {
  const int row  = lane & 15;
  const int koff = (lane >> 4) << 3;          // 0 or 8
  const float* p = base + row * ld + koff;
  float4 c0 = *reinterpret_cast<const float4*>(p);
  float4 c1 = *reinterpret_cast<const float4*>(p + 4);
  float4 c2 = *reinterpret_cast<const float4*>(p + 16);
  float4 c3 = *reinterpret_cast<const float4*>(p + 20);
  bf16x16 b;
  b[0]  = (__bf16)c0.x; b[1]  = (__bf16)c0.y; b[2]  = (__bf16)c0.z; b[3]  = (__bf16)c0.w;
  b[4]  = (__bf16)c1.x; b[5]  = (__bf16)c1.y; b[6]  = (__bf16)c1.z; b[7]  = (__bf16)c1.w;
  b[8]  = (__bf16)c2.x; b[9]  = (__bf16)c2.y; b[10] = (__bf16)c2.z; b[11] = (__bf16)c2.w;
  b[12] = (__bf16)c3.x; b[13] = (__bf16)c3.y; b[14] = (__bf16)c3.z; b[15] = (__bf16)c3.w;
  return b;
}

__device__ __forceinline__ bf16x16 fragB_lds(const __bf16* base, int ld, int lane) {
  const int row  = lane & 15;
  const int koff = (lane >> 4) << 3;
  const __bf16* p = base + row * ld + koff;
  union { uint4 q[2]; bf16x16 v; } u;
  u.q[0] = *reinterpret_cast<const uint4*>(p);
  u.q[1] = *reinterpret_cast<const uint4*>(p + 16);
  return u.v;
}

// =====================================================================
// Kernel 1: c_k = m_k + log(sum_i exp(s_ik - m_k)) per (l,h).
// grid = (8 k-slices, 512 lh). Loop order: kt outer holds bK[16] in regs
// (kt-dependent => no LICM hoist); t inner streams Q from global.
// =====================================================================
__global__ __launch_bounds__(256) void attn_stats(const float* __restrict__ Qg,
                                                  const float* __restrict__ Kg,
                                                  float* __restrict__ Cws) {
  const int ks   = blockIdx.x;     // 64-column slice
  const int lh   = blockIdx.y;
  const int kb   = ks * 64;
  const int tid  = threadIdx.x;
  const int lane = tid & 31;
  const int wave = tid >> 5;

  __shared__ alignas(16) __bf16 Ks[64 * LDP];
  __shared__ float Pm[8 * 64];
  __shared__ float Pz[8 * 64];

  const float* Qlh = Qg + (size_t)lh * SS;
  const float* Klh = Kg + (size_t)lh * SS;

  for (int idx = tid; idx < 64 * SDIM; idx += 256) {
    int r = idx >> 9;
    int c = idx & (SDIM - 1);
    Ks[r * LDP + c] = (__bf16)Klh[(size_t)(kb + r) * SDIM + c];
  }
  __syncthreads();

#pragma unroll 1
  for (int kt = 0; kt < 4; ++kt) {
    // 16-column tile of this wave's 64-column slice; hold its K fragments.
    bf16x16 bK[16];
#pragma unroll
    for (int h = 0; h < 16; ++h)
      bK[h] = fragB_lds(Ks + kt * 16 * LDP + h * 32, LDP, lane);

    float m = -1e30f, z = 0.0f;
#pragma unroll 1
    for (int t = 0; t < 4; ++t) {
      const int it = wave * 4 + t;               // 16-row i-tile
      const float* Qrow = Qlh + (size_t)it * 16 * SDIM;
      f32x8 acc = zero8();
#pragma unroll
      for (int h = 0; h < 16; ++h) {
        bf16x16 aQ = fragA_g32(Qrow + h * 32, SDIM, lane);
        acc = __builtin_amdgcn_wmma_f32_16x16x32_bf16(false, aQ, false, bK[h],
                                                      (short)0, acc, false, false);
      }
      // Online column stats. C/D layout: lane holds column n = lane&15,
      // 8 consecutive rows; lane^16 holds the other 8 rows.
      float s[8];
#pragma unroll
      for (int r = 0; r < 8; ++r) s[r] = acc[r] * SCALE;
      float tmax = s[0];
#pragma unroll
      for (int r = 1; r < 8; ++r) tmax = fmaxf(tmax, s[r]);
      tmax = fmaxf(tmax, __shfl_xor(tmax, 16, 32));
      float nm = fmaxf(m, tmax);
      float se = 0.0f;
#pragma unroll
      for (int r = 0; r < 8; ++r) se += __expf(s[r] - nm);
      se += __shfl_xor(se, 16, 32);
      z = z * __expf(m - nm) + se;
      m = nm;
    }
    if (lane < 16) {
      Pm[wave * 64 + kt * 16 + lane] = m;
      Pz[wave * 64 + kt * 16 + lane] = z;
    }
  }
  __syncthreads();

  if (tid < 64) {
    float m = Pm[tid];
#pragma unroll
    for (int w = 1; w < 8; ++w) m = fmaxf(m, Pm[w * 64 + tid]);
    float z = 0.0f;
#pragma unroll
    for (int w = 0; w < 8; ++w) z += Pz[w * 64 + tid] * __expf(Pm[w * 64 + tid] - m);
    Cws[(size_t)lh * SDIM + kb + tid] = m + __logf(z);
  }
}

// =====================================================================
// Kernel 2: 64-row O-block = exp(scale*Q K^T - c) @ V.
// grid = (8 i-blocks, 512 lh). Step 1: it outer holds aQ[16] (it-dependent,
// no hoist), kt inner streams K from global. Step 2: P @ V with V staged
// transposed in 32-row k-chunks.
// =====================================================================
__global__ __launch_bounds__(256) void attn_out(const float* __restrict__ Qg,
                                                const float* __restrict__ Kg,
                                                const float* __restrict__ Vg,
                                                const float* __restrict__ Cws,
                                                float* __restrict__ Og) {
  const int ib   = blockIdx.x;
  const int lh   = blockIdx.y;
  const int tid  = threadIdx.x;
  const int lane = tid & 31;
  const int wave = tid >> 5;
  const int n    = lane & 15;
  const int mb   = (lane >> 4) << 3;

  __shared__ alignas(16) __bf16 Qs[64 * LDP];
  __shared__ alignas(16) __bf16 Pl[64 * LDP];
  __shared__ alignas(16) __bf16 Vt[SDIM * VTP];
  __shared__ float Cl[SDIM];

  const float* Qlh = Qg + (size_t)lh * SS;
  const float* Klh = Kg + (size_t)lh * SS;
  const float* Vlh = Vg + (size_t)lh * SS;

  for (int idx = tid; idx < 64 * SDIM; idx += 256) {
    int r = idx >> 9;
    int c = idx & (SDIM - 1);
    Qs[r * LDP + c] = (__bf16)Qlh[(size_t)(ib * 64 + r) * SDIM + c];
  }
  for (int c = tid; c < SDIM; c += 256) Cl[c] = Cws[(size_t)lh * SDIM + c];
  __syncthreads();

  // ---- Step 1: P = exp(scale * Q K^T - c), wave -> 64-column k-slice ----
  const int kb = wave * 64;
#pragma unroll 1
  for (int it = 0; it < 4; ++it) {
    bf16x16 aQ[16];
#pragma unroll
    for (int h = 0; h < 16; ++h)
      aQ[h] = fragA_lds(Qs + it * 16 * LDP + h * 32, LDP, lane);
#pragma unroll 1
    for (int kt = 0; kt < 4; ++kt) {
      f32x8 acc = zero8();
#pragma unroll
      for (int h = 0; h < 16; ++h) {
        bf16x16 bK = fragB_g32(Klh + (size_t)(kb + kt * 16) * SDIM + h * 32,
                               SDIM, lane);
        acc = __builtin_amdgcn_wmma_f32_16x16x32_bf16(false, aQ[h], false, bK,
                                                      (short)0, acc, false, false);
      }
      const int kcol = kb + kt * 16 + n;
      const float cc = Cl[kcol];
#pragma unroll
      for (int r = 0; r < 8; ++r) {
        float p = __expf(acc[r] * SCALE - cc);
        Pl[(it * 16 + mb + r) * LDP + kcol] = (__bf16)p;
      }
    }
  }

  // ---- Step 2: O = P @ V, wave -> 64-column j-slice ----
  const int jb = wave * 64;
  f32x8 o[4][4];
#pragma unroll
  for (int it = 0; it < 4; ++it)
#pragma unroll
    for (int jt = 0; jt < 4; ++jt) o[it][jt] = zero8();

#pragma unroll 1
  for (int kc = 0; kc < 16; ++kc) {
    __syncthreads();
    // Stage V rows [kc*32, kc*32+32) transposed: Vt[j][kk] = V[kc*32+kk][j]
    for (int idx = tid; idx < 32 * SDIM; idx += 256) {
      int kk = idx >> 9;
      int j  = idx & (SDIM - 1);
      Vt[j * VTP + kk] = (__bf16)Vlh[(size_t)(kc * 32 + kk) * SDIM + j];
    }
    __syncthreads();
    bf16x16 bV[4];
#pragma unroll
    for (int jt = 0; jt < 4; ++jt)
      bV[jt] = fragB_lds(Vt + (jb + jt * 16) * VTP, VTP, lane);
#pragma unroll
    for (int it = 0; it < 4; ++it) {
      bf16x16 aP = fragA_lds(Pl + it * 16 * LDP + kc * 32, LDP, lane);
#pragma unroll
      for (int jt = 0; jt < 4; ++jt) {
        o[it][jt] = __builtin_amdgcn_wmma_f32_16x16x32_bf16(false, aP, false, bV[jt],
                                                            (short)0, o[it][jt],
                                                            false, false);
      }
    }
  }

  float* Olh = Og + (size_t)lh * SS;
#pragma unroll
  for (int it = 0; it < 4; ++it) {
#pragma unroll
    for (int jt = 0; jt < 4; ++jt) {
#pragma unroll
      for (int r = 0; r < 8; ++r) {
        Olh[(size_t)(ib * 64 + it * 16 + mb + r) * SDIM + jb + jt * 16 + n] =
            o[it][jt][r];
      }
    }
  }
}

extern "C" void kernel_launch(void* const* d_in, const int* in_sizes, int n_in,
                              void* d_out, int out_size, void* d_ws, size_t ws_size,
                              hipStream_t stream) {
  // d_in[0] = x1 (unused by the reference), [1]=query, [2]=key, [3]=value (fp32)
  const float* q = (const float*)d_in[1];
  const float* k = (const float*)d_in[2];
  const float* v = (const float*)d_in[3];
  float* out = (float*)d_out;
  float* cws = (float*)d_ws;        // 512*512 floats = 1 MB of scratch

  dim3 blk(256);
  attn_stats<<<dim3(8, 512), blk, 0, stream>>>(q, k, cws);
  attn_out<<<dim3(8, 512), blk, 0, stream>>>(q, k, v, cws, out);
}
